// TransformerBlock_24584392802334
// MI455X (gfx1250) — matrix operand order repaired and checked by reading, not compiled, not run on hardware
//
#include <hip/hip_runtime.h>

// ---------------------------------------------------------------------------
// PointTransformer-style graph attention layer for MI455X (gfx1250, wave32).
// All 64x64 projections run on v_wmma_f32_16x16x32_bf16 (f32 accumulate).
// Fragments are fed with ds_load_b128: A-tiles row-major, weights transposed
// (N-major) in LDS so every lane's B-fragment is two 8-elem contiguous runs.
// Row strides padded to 72 elems (144B) to spread LDS banks, 16B aligned.
// ---------------------------------------------------------------------------

typedef __attribute__((ext_vector_type(16))) __bf16 v16bf;
typedef __attribute__((ext_vector_type(8)))  __bf16 v8bf;
typedef __attribute__((ext_vector_type(2)))  __bf16 v2bf;
typedef __attribute__((ext_vector_type(8)))  float  v8f;
typedef __attribute__((ext_vector_type(2)))  float  v2f;

#define CDIM 64
#define TSTRIDE 72  // LDS tile row stride in bf16 elems (144B: 16B aligned)
#define WSTRIDE 72  // LDS transposed-weight row stride

// Intra-wave LDS write -> read ordering (CDNA5 split wait counters).
__device__ inline void lds_fence() {
  asm volatile("s_wait_dscnt 0x0" ::: "memory");
}

__device__ inline v16bf cat8(v8bf lo, v8bf hi) {
  return __builtin_shufflevector(lo, hi, 0, 1, 2, 3, 4, 5, 6, 7, 8, 9, 10, 11,
                                 12, 13, 14, 15);
}

// A-fragment (16x32 bf16, K chunk at k0) from a row-major LDS tile.
// ISA 7.12.2: lane m=lane&15; lanes<16 hold K kb+0..7 / kb+16..23 with kb=k0,
// lanes>=16 the +8 variants. Both halves are contiguous -> 2x ds_load_b128.
__device__ inline v16bf load_a_frag(const __bf16* T, int lane, int k0) {
  int m  = lane & 15;
  int kb = k0 + ((lane & 16) ? 8 : 0);
  const __bf16* p = T + m * TSTRIDE + kb;
  v8bf lo = *(const v8bf*)(p);
  v8bf hi = *(const v8bf*)(p + 16);
  return cat8(lo, hi);
}

// B-fragment (32x16 bf16, chunk k0,n0) from an N-major (transposed) LDS
// weight: lane n holds K=kb..kb+7 / kb+16..kb+23 of column n (kb=k0, +8 for
// lanes>=16), matching the A-fragment K order.
__device__ inline v16bf load_b_frag(const __bf16* Wt, int lane, int k0, int n0) {
  int n  = n0 + (lane & 15);
  int kb = k0 + ((lane & 16) ? 8 : 0);
  const __bf16* p = Wt + n * WSTRIDE + kb;
  v8bf lo = *(const v8bf*)(p);
  v8bf hi = *(const v8bf*)(p + 16);
  return cat8(lo, hi);
}

// D(16x64) = A(16x64) @ W(64x64): 4 N-tiles x 2 K-chunks = 8 WMMAs.
__device__ inline void wave_gemm(const __bf16* At, const __bf16* Wt, int lane,
                                 v8f acc[4]) {
  v16bf a0 = load_a_frag(At, lane, 0);
  v16bf a1 = load_a_frag(At, lane, 32);
#pragma unroll
  for (int nt = 0; nt < 4; ++nt) {
    v8f c = {};
    c = __builtin_amdgcn_wmma_f32_16x16x32_bf16(
        false, a0, false, load_b_frag(Wt, lane, 0, nt * 16), (short)0, c,
        false, false);
    c = __builtin_amdgcn_wmma_f32_16x16x32_bf16(
        false, a1, false, load_b_frag(Wt, lane, 32, nt * 16), (short)0, c,
        false, false);
    acc[nt] = c;
  }
}

// Cooperative f32 -> bf16 transpose of a 64x64 weight into LDS (N-major).
__device__ inline void stage_weight(__bf16* Wt, const float* W, int tid) {
  for (int i = tid; i < 4096; i += 256) {
    int k = i >> 6, n = i & 63;  // source read coalesced (row-major)
    Wt[n * WSTRIDE + k] = (__bf16)W[i];
  }
}

// ---------------------------------------------------------------------------
// Kernel 0: zero amax (0.0f == bits 0 -> valid uint atomicMax identity for
// non-negative floats), denom and numerator accumulators.
// ---------------------------------------------------------------------------
__global__ void k_init(float* __restrict__ amax, float* __restrict__ denom,
                       float* __restrict__ num, long long n) {
  long long i      = (long long)blockIdx.x * blockDim.x + threadIdx.x;
  long long stride = (long long)gridDim.x * blockDim.x;
  for (; i < n; i += stride) {
    amax[i]  = 0.f;
    denom[i] = 0.f;
    num[i]   = 0.f;
  }
}

// ---------------------------------------------------------------------------
// Kernel 1: node projections. x1 = relu(x@W_in+b_in); v = x1@W_lin;
// a_src = x1@W_src; a_dst = x1@W_dst.  8 waves/block, 16 nodes/wave.
// ---------------------------------------------------------------------------
__global__ __launch_bounds__(256) void k_node(
    const float* __restrict__ x, const float* __restrict__ Win,
    const float* __restrict__ bin, const float* __restrict__ Wlin,
    const float* __restrict__ Wsrc, const float* __restrict__ Wdst,
    float* __restrict__ vproj, float* __restrict__ asrc,
    float* __restrict__ adst, int N) {
  __shared__ __bf16 sW[4][CDIM * WSTRIDE];
  __shared__ __bf16 sT[8][16 * TSTRIDE];
  int tid = threadIdx.x;
  stage_weight(sW[0], Win, tid);
  stage_weight(sW[1], Wlin, tid);
  stage_weight(sW[2], Wsrc, tid);
  stage_weight(sW[3], Wdst, tid);
  __syncthreads();

  int wave = tid >> 5, lane = tid & 31;
  int row0  = (blockIdx.x * 8 + wave) * 16;
  __bf16* T = sT[wave];

  // load x tile (16x64) -> bf16 LDS, coalesced
#pragma unroll
  for (int i = 0; i < 32; ++i) {
    int idx = lane + i * 32;
    int r = idx >> 6, c = idx & 63;
    int gr   = row0 + r;
    float xv = (gr < N) ? x[(size_t)gr * CDIM + c] : 0.f;
    T[r * TSTRIDE + c] = (__bf16)xv;
  }
  lds_fence();

  v8f acc[4];
  wave_gemm(T, sW[0], lane, acc);  // x @ W_in

  // x1 = relu(acc + b_in) -> overwrite tile (DS ops are wave-ordered)
#pragma unroll
  for (int nt = 0; nt < 4; ++nt) {
    int col  = nt * 16 + (lane & 15);
    float bb = bin[col];
#pragma unroll
    for (int r = 0; r < 8; ++r) {
      int m = r + ((lane & 16) ? 8 : 0);
      T[m * TSTRIDE + col] = (__bf16)fmaxf(acc[nt][r] + bb, 0.f);
    }
  }
  lds_fence();

  // reuse A-fragments for the three bias-free projections
  v16bf a0 = load_a_frag(T, lane, 0);
  v16bf a1 = load_a_frag(T, lane, 32);
  float* outs[3] = {vproj, asrc, adst};
#pragma unroll
  for (int wsel = 0; wsel < 3; ++wsel) {
    const __bf16* Wt = sW[wsel + 1];
    float* dst       = outs[wsel];
#pragma unroll
    for (int nt = 0; nt < 4; ++nt) {
      v8f c = {};
      c = __builtin_amdgcn_wmma_f32_16x16x32_bf16(
          false, a0, false, load_b_frag(Wt, lane, 0, nt * 16), (short)0, c,
          false, false);
      c = __builtin_amdgcn_wmma_f32_16x16x32_bf16(
          false, a1, false, load_b_frag(Wt, lane, 32, nt * 16), (short)0, c,
          false, false);
      int col = nt * 16 + (lane & 15);
#pragma unroll
      for (int r = 0; r < 8; ++r) {
        int m  = r + ((lane & 16) ? 8 : 0);
        int gr = row0 + m;
        if (gr < N) dst[(size_t)gr * CDIM + col] = c[r];
      }
    }
  }
}

// ---------------------------------------------------------------------------
// Kernel 2: per-edge MLPs. delta = mlp2(pos_i - pos_j); alpha = mlp2(a_i - a_j
// + delta). Stores delta/alpha as bf16 streams, atomic-max alpha into amax
// (alpha >= 0 after final ReLU -> raw-bit uint max is exact).
// ---------------------------------------------------------------------------
__global__ __launch_bounds__(256) void k_edge1(
    const float* __restrict__ pos, const int* __restrict__ eidx,
    const float* __restrict__ asrc, const float* __restrict__ adst,
    const float* __restrict__ posW1, const float* __restrict__ posb1,
    const float* __restrict__ posW2, const float* __restrict__ posb2,
    const float* __restrict__ attW1, const float* __restrict__ attb1,
    const float* __restrict__ attW2, const float* __restrict__ attb2,
    __bf16* __restrict__ delta_bf, __bf16* __restrict__ alpha_bf,
    unsigned int* __restrict__ amax_u, int E) {
  __shared__ __bf16 sW[3][CDIM * WSTRIDE];
  __shared__ __bf16 sT[8][16 * TSTRIDE];
  int tid = threadIdx.x;
  stage_weight(sW[0], posW2, tid);
  stage_weight(sW[1], attW1, tid);
  stage_weight(sW[2], attW2, tid);
  __syncthreads();

  int wave = tid >> 5, lane = tid & 31;
  int e0    = (blockIdx.x * 8 + wave) * 16;
  __bf16* T = sT[wave];
  int c0 = lane, c1 = lane + 32;

  // pos MLP layer 1 (3 -> 64), scalar FMAs (K=3 too small for WMMA)
#pragma unroll 1
  for (int e = 0; e < 16; ++e) {
    int eg = min(e0 + e, E - 1);
    int sj = eidx[eg], dj = eidx[E + eg];
    float d0 = pos[dj * 3 + 0] - pos[sj * 3 + 0];
    float d1 = pos[dj * 3 + 1] - pos[sj * 3 + 1];
    float d2 = pos[dj * 3 + 2] - pos[sj * 3 + 2];
    float h0 = fmaxf(posb1[c0] + d0 * posW1[c0] + d1 * posW1[64 + c0] +
                         d2 * posW1[128 + c0], 0.f);
    float h1 = fmaxf(posb1[c1] + d0 * posW1[c1] + d1 * posW1[64 + c1] +
                         d2 * posW1[128 + c1], 0.f);
    T[e * TSTRIDE + c0] = (__bf16)h0;
    T[e * TSTRIDE + c1] = (__bf16)h1;
  }
  lds_fence();

  v8f acc[4];
  wave_gemm(T, sW[0], lane, acc);  // h1 @ pos_W2

  // delta = relu(acc + posb2): store bf16, build ain = a_dst - a_src + delta
#pragma unroll
  for (int nt = 0; nt < 4; ++nt) {
    int col  = nt * 16 + (lane & 15);
    float bb = posb2[col];
#pragma unroll
    for (int r = 0; r < 8; ++r) {
      int m    = r + ((lane & 16) ? 8 : 0);
      int eg   = e0 + m;
      bool ok  = eg < E;
      int egc  = ok ? eg : (E - 1);
      float dv = fmaxf(acc[nt][r] + bb, 0.f);
      if (ok) delta_bf[(size_t)eg * CDIM + col] = (__bf16)dv;
      int sj = eidx[egc], dj = eidx[E + egc];
      float ain = adst[(size_t)dj * CDIM + col] -
                  asrc[(size_t)sj * CDIM + col] + dv;
      T[m * TSTRIDE + col] = (__bf16)ain;
    }
  }
  lds_fence();

  wave_gemm(T, sW[1], lane, acc);  // att layer 1
#pragma unroll
  for (int nt = 0; nt < 4; ++nt) {
    int col  = nt * 16 + (lane & 15);
    float bb = attb1[col];
#pragma unroll
    for (int r = 0; r < 8; ++r) {
      int m = r + ((lane & 16) ? 8 : 0);
      T[m * TSTRIDE + col] = (__bf16)fmaxf(acc[nt][r] + bb, 0.f);
    }
  }
  lds_fence();

  wave_gemm(T, sW[2], lane, acc);  // att layer 2 -> alpha
#pragma unroll
  for (int nt = 0; nt < 4; ++nt) {
    int col  = nt * 16 + (lane & 15);
    float bb = attb2[col];
#pragma unroll
    for (int r = 0; r < 8; ++r) {
      int m  = r + ((lane & 16) ? 8 : 0);
      int eg = e0 + m;
      if (eg < E) {
        float av = fmaxf(acc[nt][r] + bb, 0.f);
        alpha_bf[(size_t)eg * CDIM + col] = (__bf16)av;
        int dj = eidx[E + eg];
        atomicMax(&amax_u[(size_t)dj * CDIM + col], __float_as_uint(av));
      }
    }
  }
}

// ---------------------------------------------------------------------------
// Kernel 3: ex = exp(alpha - amax[dst]); scatter-add denom and ex*(v+delta)
// into L2-resident node accumulators (global_atomic_add_f32). Each thread
// owns an adjacent channel pair -> b32/b64 coalesced loads.
// ---------------------------------------------------------------------------
__global__ __launch_bounds__(256) void k_edge2(
    const int* __restrict__ eidx, const float* __restrict__ vproj,
    const float* __restrict__ amax, const __bf16* __restrict__ delta_bf,
    const __bf16* __restrict__ alpha_bf, float* __restrict__ denom,
    float* __restrict__ num, int E) {
  long long gid = (long long)blockIdx.x * blockDim.x + threadIdx.x;
  long long e   = gid >> 5;
  if (e >= (long long)E) return;
  int lane     = (int)(gid & 31);
  int c        = lane * 2;
  int sj       = eidx[e];
  int dj       = eidx[(size_t)E + e];
  size_t ebase = (size_t)e * CDIM + c;
  size_t dbase = (size_t)dj * CDIM + c;
  size_t sbase = (size_t)sj * CDIM + c;

  v2bf ap = *(const v2bf*)(alpha_bf + ebase);
  v2bf dp = *(const v2bf*)(delta_bf + ebase);
  v2f  am = *(const v2f*)(amax + dbase);
  v2f  vv = *(const v2f*)(vproj + sbase);
#pragma unroll
  for (int h = 0; h < 2; ++h) {
    float ex = __expf((float)ap[h] - am[h]);
    atomicAdd(&denom[dbase + h], ex);
    atomicAdd(&num[dbase + h], ex * (vv[h] + (float)dp[h]));
  }
}

// ---------------------------------------------------------------------------
// Kernel 4: out = relu((num/(denom+1e-16)) @ W_out + b_out)
// ---------------------------------------------------------------------------
__global__ __launch_bounds__(256) void k_out(
    const float* __restrict__ num, const float* __restrict__ denom,
    const float* __restrict__ Wout, const float* __restrict__ bout,
    float* __restrict__ out, int N) {
  __shared__ __bf16 sW[CDIM * WSTRIDE];
  __shared__ __bf16 sT[8][16 * TSTRIDE];
  int tid = threadIdx.x;
  stage_weight(sW, Wout, tid);
  __syncthreads();

  int wave = tid >> 5, lane = tid & 31;
  int row0  = (blockIdx.x * 8 + wave) * 16;
  __bf16* T = sT[wave];
#pragma unroll
  for (int i = 0; i < 32; ++i) {
    int idx = lane + i * 32;
    int r = idx >> 6, c = idx & 63;
    int gr  = row0 + r;
    float h = 0.f;
    if (gr < N) {
      size_t o = (size_t)gr * CDIM + c;
      h        = num[o] / (denom[o] + 1e-16f);
    }
    T[r * TSTRIDE + c] = (__bf16)h;
  }
  lds_fence();

  v8f acc[4];
  wave_gemm(T, sW, lane, acc);
#pragma unroll
  for (int nt = 0; nt < 4; ++nt) {
    int col  = nt * 16 + (lane & 15);
    float bb = bout[col];
#pragma unroll
    for (int r = 0; r < 8; ++r) {
      int m  = r + ((lane & 16) ? 8 : 0);
      int gr = row0 + m;
      if (gr < N) out[(size_t)gr * CDIM + col] = fmaxf(acc[nt][r] + bb, 0.f);
    }
  }
}

// ---------------------------------------------------------------------------
extern "C" void kernel_launch(void* const* d_in, const int* in_sizes, int n_in,
                              void* d_out, int out_size, void* d_ws,
                              size_t ws_size, hipStream_t stream) {
  (void)n_in; (void)out_size; (void)ws_size;
  const float* x     = (const float*)d_in[0];
  const float* pos   = (const float*)d_in[1];
  const float* Win   = (const float*)d_in[2];
  const float* bin   = (const float*)d_in[3];
  const float* Wlin  = (const float*)d_in[4];
  const float* Wsrc  = (const float*)d_in[5];
  const float* Wdst  = (const float*)d_in[6];
  const float* posW1 = (const float*)d_in[7];
  const float* posb1 = (const float*)d_in[8];
  const float* posW2 = (const float*)d_in[9];
  const float* posb2 = (const float*)d_in[10];
  const float* attW1 = (const float*)d_in[11];
  const float* attb1 = (const float*)d_in[12];
  const float* attW2 = (const float*)d_in[13];
  const float* attb2 = (const float*)d_in[14];
  const float* Wout  = (const float*)d_in[15];
  const float* bout  = (const float*)d_in[16];
  const int* eidx    = (const int*)d_in[17];
  int N = in_sizes[0] / CDIM;
  int E = in_sizes[17] / 2;
  float* out = (float*)d_out;

  char* ws   = (char*)d_ws;
  size_t off = 0;
  auto alloc = [&](size_t bytes) -> void* {
    void* p = ws + off;
    off += (bytes + 255) & ~(size_t)255;
    return p;
  };
  float* vproj   = (float*)alloc((size_t)N * CDIM * 4);
  float* asrc    = (float*)alloc((size_t)N * CDIM * 4);
  float* adst    = (float*)alloc((size_t)N * CDIM * 4);
  float* amax    = (float*)alloc((size_t)N * CDIM * 4);
  float* denom   = (float*)alloc((size_t)N * CDIM * 4);
  float* numb    = (float*)alloc((size_t)N * CDIM * 4);
  __bf16* deltab = (__bf16*)alloc((size_t)E * CDIM * 2);
  __bf16* alphab = (__bf16*)alloc((size_t)E * CDIM * 2);

  k_init<<<2048, 256, 0, stream>>>(amax, denom, numb, (long long)N * CDIM);

  int nodeBlocks = (N + 127) / 128;
  k_node<<<nodeBlocks, 256, 0, stream>>>(x, Win, bin, Wlin, Wsrc, Wdst, vproj,
                                         asrc, adst, N);

  int edgeBlocks = (E + 127) / 128;
  k_edge1<<<edgeBlocks, 256, 0, stream>>>(pos, eidx, asrc, adst, posW1, posb1,
                                          posW2, posb2, attW1, attb1, attW2,
                                          attb2, deltab, alphab,
                                          (unsigned int*)amax, E);

  long long e2threads = (long long)E * 32;
  int e2blocks        = (int)((e2threads + 255) / 256);
  k_edge2<<<e2blocks, 256, 0, stream>>>(eidx, vproj, amax, deltab, alphab,
                                        denom, numb, E);

  k_out<<<nodeBlocks, 256, 0, stream>>>(numb, denom, Wout, bout, out, N);
}
